// CAMoE_GNN_7086696038966
// MI455X (gfx1250) — matrix operand order repaired and checked by reading, not compiled, not run on hardware
//
#include <hip/hip_runtime.h>
#include <hip/hip_bf16.h>
#include <stdint.h>

typedef __bf16 bf16;
typedef __attribute__((ext_vector_type(16))) __bf16 v16bf;
typedef __attribute__((ext_vector_type(8)))  float  v8f;

#define FDIM 128          // F_IN == HID
#define EXPN 3
#define OUTN 64
#define TEMP_INV (1.0f/101.0f)

// ---------------- degree / norm ----------------
__global__ __launch_bounds__(256) void k_deg_init(float* deg, int N) {
  int i = blockIdx.x * 256 + threadIdx.x;
  if (i < N) deg[i] = 1.0f;                      // self-loop contributes 1
}

__global__ __launch_bounds__(256) void k_deg_edges(const int* __restrict__ ei,
                                                   float* deg, int E) {
  int i = blockIdx.x * 256 + threadIdx.x;
  if (i < E) atomicAdd(&deg[ei[E + i]], 1.0f);   // dst row of edge_index
}

__global__ __launch_bounds__(256) void k_rsqrt_inplace(float* d, int N) {
  int i = blockIdx.x * 256 + threadIdx.x;
  if (i < N) d[i] = rsqrtf(d[i]);
}

// ---------------- gates (both layers at once) ----------------
__global__ __launch_bounds__(256) void k_gates(const float* __restrict__ top,
                                               const float* __restrict__ Wg0,
                                               const float* __restrict__ Wg1,
                                               float* g0, float* g1, int N) {
  int n = blockIdx.x * 256 + threadIdx.x;
  if (n >= N) return;
  float t0 = top[n*4+0], t1 = top[n*4+1], t2 = top[n*4+2], t3 = top[n*4+3];
  const float* wgs[2] = { Wg0, Wg1 };
  float* outs[2] = { g0, g1 };
  #pragma unroll
  for (int L = 0; L < 2; ++L) {
    const float* wg = wgs[L];
    float l[EXPN];
    #pragma unroll
    for (int e = 0; e < EXPN; ++e)
      l[e] = (t0*wg[e*4+0] + t1*wg[e*4+1] + t2*wg[e*4+2] + t3*wg[e*4+3]) * TEMP_INV;
    float m = fmaxf(l[0], fmaxf(l[1], l[2]));
    float ex[EXPN], s = 0.f;
    #pragma unroll
    for (int e = 0; e < EXPN; ++e) { ex[e] = __expf(l[e] - m); s += ex[e]; }
    float inv = 1.0f / s;
    #pragma unroll
    for (int e = 0; e < EXPN; ++e) outs[L][n*EXPN + e] = ex[e] * inv;
  }
}

// ---------------- weight prep ----------------
// fp32 W[e][k][n] -> fragment-packed bf16 WtF[e][nt][kt][lane][elem]
// B-fragment layout (32x16 bf16, matches A-fragment k order):
//   lanes 0-15  = col n, K k0+0..7 (elems 0..7), k0+16..23 (elems 8..15)
//   lanes 16-31 = col n, K k0+8..15 (elems 0..7), k0+24..31 (elems 8..15)
__global__ __launch_bounds__(256) void k_wprep(const float* __restrict__ W,
                                               bf16* __restrict__ WtF) {
  int idx = blockIdx.x * 256 + threadIdx.x;           // over 3*128*128, source order
  if (idx >= EXPN * FDIM * FDIM) return;
  int e = idx >> 14;
  int rem = idx & 16383;
  int k = rem >> 7;
  int n = rem & 127;
  int nt = n >> 4, ncol = n & 15;
  int kt = k >> 5, dk = k & 31;
  int lane = ncol + 16 * ((dk >> 3) & 1);
  int elem = (dk & 7) + 8 * (dk >> 4);
  WtF[((((e * 8 + nt) * 4) + kt) * 32 + lane) * 16 + elem] = (bf16)W[idx];
}

// ---------------- hagg = Ahat * h : self-loop init + edge scatter ----------------
__global__ __launch_bounds__(256) void k_selfloop(const float* __restrict__ h,
                                                  const float* __restrict__ dinv,
                                                  float* __restrict__ hagg, int total) {
  int i = blockIdx.x * 256 + threadIdx.x;             // over N*128
  if (i < total) {
    float dv = dinv[i >> 7];
    hagg[i] = h[i] * dv * dv;
  }
}

__global__ __launch_bounds__(256) void k_scatter(const float* __restrict__ h,
                                                 const int* __restrict__ ei,
                                                 const float* __restrict__ dinv,
                                                 float* __restrict__ hagg, int E) {
  int t = blockIdx.x * 256 + threadIdx.x;             // E*32 lanes
  int e = t >> 5;
  if (e >= E) return;
  int lane = t & 31;
  int s = ei[e];
  int d = ei[E + e];
  float nrm = dinv[s] * dinv[d];
  const float4 v = *(const float4*)(h + (size_t)s * FDIM + lane * 4);
  float* o = hagg + (size_t)d * FDIM + lane * 4;
  atomicAdd(o + 0, v.x * nrm);
  atomicAdd(o + 1, v.y * nrm);
  atomicAdd(o + 2, v.z * nrm);
  atomicAdd(o + 3, v.w * nrm);
}

// ---------------- fused 3-expert WMMA GEMM ----------------
// out[n] = sum_e gate[n][e] * relu(hagg[n] @ W[e] + b[e])
// Block = 256 threads = 8 waves; block tile = 128 rows; wave tile = 16 rows x 128 cols.
// A is staged in LDS in fragment-packed order; B fragments are coalesced 1KB
// blocks in global (L2-resident, 96KB total).
__global__ __launch_bounds__(256) void k_expert_gemm(
    const float* __restrict__ hagg,   // N x 128 fp32
    const bf16*  __restrict__ WtF,    // fragment-packed, 3*8*4*32*16 bf16
    const float* __restrict__ bias,   // 3 x 128
    const float* __restrict__ gate,   // N x 3
    float* __restrict__ out,          // N x 128 fp32
    int N)
{
  __shared__ __align__(32) bf16 sA[8 * 4 * 32 * 16];  // 32 KB, fragment-packed
  __shared__ float sGate[128 * EXPN];                 // [local row][e]
  __shared__ float sBias[EXPN * FDIM];                // [e][n]

  const int tid  = threadIdx.x;
  const int wave = tid >> 5;
  const int lane = tid & 31;
  const int blockRow = blockIdx.x * 128;

  // stage bias + gates
  for (int i = tid; i < EXPN * FDIM; i += 256) sBias[i] = bias[i];
  for (int i = tid; i < 128 * EXPN; i += 256) {
    int gr = blockRow + i / EXPN;
    sGate[i] = (gr < N) ? gate[(size_t)blockRow * EXPN + i] : 0.f;
  }

  // Stage A tile fp32 -> bf16 directly into A-fragment layout.
  // A-fragment (16x32 bf16 per ISA): for local row r = w*16+m, K index k = kt*32+dk:
  //   lane = m + 16*((dk>>3)&1) ; elem = (dk&7) + 8*(dk>>4)
  for (int i = tid; i < 128 * 32; i += 256) {
    int r  = i >> 5;               // 0..127
    int c4 = (i & 31) << 2;        // k = c4 .. c4+3 (same elem group)
    int gr = blockRow + r;
    float4 v = make_float4(0.f, 0.f, 0.f, 0.f);
    if (gr < N) v = *(const float4*)(hagg + (size_t)gr * FDIM + c4);
    int w = r >> 4, m = r & 15;
    int kt = c4 >> 5, dk = c4 & 31;
    int laneA = m + 16 * ((dk >> 3) & 1);
    int elem  = (dk & 7) + 8 * (dk >> 4);
    bf16* dst = sA + (((w * 4 + kt) * 32 + laneA) * 16 + elem);
    dst[0] = (bf16)v.x; dst[1] = (bf16)v.y; dst[2] = (bf16)v.z; dst[3] = (bf16)v.w;
  }
  __syncthreads();

  // A fragments: one contiguous 32B ds load per kt
  v16bf afrag[4];
  #pragma unroll
  for (int kt = 0; kt < 4; ++kt)
    afrag[kt] = *(const v16bf*)(sA + ((wave * 4 + kt) * 32 + lane) * 16);

  // C/D element i lives at local row mrow+i, col nt*16+ncol
  const int mrow = wave * 16 + ((lane >> 4) * 8);
  const int ncol = lane & 15;
  const int gbase = blockRow + mrow;

  float g[EXPN][8];
  #pragma unroll
  for (int e = 0; e < EXPN; ++e)
    #pragma unroll
    for (int i = 0; i < 8; ++i)
      g[e][i] = sGate[(mrow + i) * EXPN + e];

  #pragma unroll
  for (int nt = 0; nt < 8; ++nt) {
    v8f oacc = (v8f){0.f,0.f,0.f,0.f,0.f,0.f,0.f,0.f};
    #pragma unroll
    for (int e = 0; e < EXPN; ++e) {
      v8f acc = (v8f){0.f,0.f,0.f,0.f,0.f,0.f,0.f,0.f};
      #pragma unroll
      for (int kt = 0; kt < 4; ++kt) {
        // coalesced: 32 lanes x 32B = one contiguous 1KB fragment block
        v16bf bfrag = *(const v16bf*)(WtF +
            (((size_t)((e * 8 + nt) * 4 + kt)) * 32 + lane) * 16);
        acc = __builtin_amdgcn_wmma_f32_16x16x32_bf16(
                  false, afrag[kt], false, bfrag, (short)0, acc, false, false);
      }
      float bv = sBias[e * FDIM + nt * 16 + ncol];
      #pragma unroll
      for (int i = 0; i < 8; ++i) {
        float v = acc[i] + bv;
        v = v > 0.f ? v : 0.f;
        oacc[i] += g[e][i] * v;
      }
    }
    #pragma unroll
    for (int i = 0; i < 8; ++i) {
      int gr = gbase + i;
      if (gr < N) out[(size_t)gr * FDIM + nt * 16 + ncol] = oacc[i];
    }
  }
}

// ---------------- pooling + final dense ----------------
__global__ __launch_bounds__(256) void k_pool_zero(float* sums, float* cnt, int G) {
  int i = blockIdx.x * 256 + threadIdx.x;
  if (i < G * FDIM) sums[i] = 0.f;
  if (i < G) cnt[i] = 0.f;
}

__global__ __launch_bounds__(256) void k_pool_acc(const float* __restrict__ h,
                                                  const int* __restrict__ batch,
                                                  float* sums, float* cnt, int N) {
  int t = blockIdx.x * 256 + threadIdx.x;
  int n = t >> 5;
  if (n >= N) return;
  int lane = t & 31;
  int b = batch[n];
  const float4 v = *(const float4*)(h + (size_t)n * FDIM + lane * 4);
  float* o = sums + (size_t)b * FDIM + lane * 4;
  atomicAdd(o + 0, v.x);
  atomicAdd(o + 1, v.y);
  atomicAdd(o + 2, v.z);
  atomicAdd(o + 3, v.w);
  if (lane == 0) atomicAdd(&cnt[b], 1.0f);
}

__global__ __launch_bounds__(64) void k_final(const float* __restrict__ sums,
                                              const float* __restrict__ cnt,
                                              const float* __restrict__ Wf,
                                              const float* __restrict__ bfv,
                                              float* __restrict__ out, int G) {
  __shared__ float pooled[FDIM];
  int g = blockIdx.x;
  int o = threadIdx.x;
  float inv = 1.0f / fmaxf(cnt[g], 1.0f);
  for (int k = o; k < FDIM; k += 64) pooled[k] = sums[g * FDIM + k] * inv;
  __syncthreads();
  float acc = bfv[o];
  for (int k = 0; k < FDIM; ++k) acc += pooled[k] * Wf[k * OUTN + o];
  out[g * OUTN + o] = acc;
}

// ---------------- launch ----------------
static inline size_t alignup(size_t x) { return (x + 255) & ~(size_t)255; }

extern "C" void kernel_launch(void* const* d_in, const int* in_sizes, int n_in,
                              void* d_out, int out_size, void* d_ws, size_t ws_size,
                              hipStream_t stream) {
  (void)n_in; (void)ws_size;
  const float* x    = (const float*)d_in[0];
  const float* top  = (const float*)d_in[1];
  const int*   ei   = (const int*)  d_in[2];
  const int*   batch= (const int*)  d_in[3];
  const float* W0   = (const float*)d_in[4];
  const float* b0   = (const float*)d_in[5];
  const float* Wg0  = (const float*)d_in[6];
  const float* W1   = (const float*)d_in[7];
  const float* b1   = (const float*)d_in[8];
  const float* Wg1  = (const float*)d_in[9];
  const float* Wf   = (const float*)d_in[10];
  const float* bfv  = (const float*)d_in[11];
  float* out = (float*)d_out;

  const int N = in_sizes[0] / FDIM;
  const int E = in_sizes[2] / 2;
  const int G = out_size / OUTN;

  // scratch carve-out
  char* w = (char*)d_ws;
  float* dinv  = (float*)w; w += alignup((size_t)N * 4);
  float* gate0 = (float*)w; w += alignup((size_t)N * EXPN * 4);
  float* gate1 = (float*)w; w += alignup((size_t)N * EXPN * 4);
  float* bufA  = (float*)w; w += alignup((size_t)N * FDIM * 4);
  float* bufB  = (float*)w; w += alignup((size_t)N * FDIM * 4);
  float* sums  = (float*)w; w += alignup((size_t)G * FDIM * 4);
  float* cnt   = (float*)w; w += alignup((size_t)G * 4);
  bf16*  Wt0   = (bf16*)w;  w += alignup((size_t)EXPN * FDIM * FDIM * 2);
  bf16*  Wt1   = (bf16*)w;

  dim3 blk(256);
  int gN    = (N + 255) / 256;
  int gE    = (E + 255) / 256;
  int gNE   = ((int)((size_t)N * FDIM) + 255) / 256;
  int gE32  = (int)(((size_t)E * 32 + 255) / 256);
  int gN32  = (int)(((size_t)N * 32 + 255) / 256);
  int gW    = (EXPN * FDIM * FDIM + 255) / 256;
  int gGemm = (N + 127) / 128;
  int gPool = (G * FDIM + 255) / 256;

  // degree / symmetric norm
  k_deg_init<<<gN, blk, 0, stream>>>(dinv, N);
  k_deg_edges<<<gE, blk, 0, stream>>>(ei, dinv, E);
  k_rsqrt_inplace<<<gN, blk, 0, stream>>>(dinv, N);

  // gates + fragment-packed bf16 weights
  k_gates<<<gN, blk, 0, stream>>>(top, Wg0, Wg1, gate0, gate1, N);
  k_wprep<<<gW, blk, 0, stream>>>(W0, Wt0);
  k_wprep<<<gW, blk, 0, stream>>>(W1, Wt1);

  // layer 1: hagg = Ahat * x ; h1 = MoE(hagg)
  k_selfloop<<<gNE, blk, 0, stream>>>(x, dinv, bufA, N * FDIM);
  k_scatter<<<gE32, blk, 0, stream>>>(x, ei, dinv, bufA, E);
  k_expert_gemm<<<gGemm, blk, 0, stream>>>(bufA, Wt0, b0, gate0, bufB, N);

  // layer 2: hagg = Ahat * h1 ; h2 = MoE(hagg)
  k_selfloop<<<gNE, blk, 0, stream>>>(bufB, dinv, bufA, N * FDIM);
  k_scatter<<<gE32, blk, 0, stream>>>(bufB, ei, dinv, bufA, E);
  k_expert_gemm<<<gGemm, blk, 0, stream>>>(bufA, Wt1, b1, gate1, bufB, N);

  // mean pool + final dense
  k_pool_zero<<<gPool, blk, 0, stream>>>(sums, cnt, G);
  k_pool_acc<<<gN32, blk, 0, stream>>>(bufB, batch, sums, cnt, N);
  k_final<<<G, dim3(64), 0, stream>>>(sums, cnt, Wf, bfv, out, G);
}
